// SelectiveStateSpaceCell_83958020702584
// MI455X (gfx1250) — matrix operand order repaired and checked by reading, not compiled, not run on hardware
//
#include <hip/hip_runtime.h>
#include <hip/hip_bf16.h>

// ---------------------------------------------------------------------------
// Problem constants
// ---------------------------------------------------------------------------
#define BB 16384
#define DD 1024
#define SS 2048
#define LN_EPS 1e-5f

typedef __bf16 bf16_t;
typedef __attribute__((ext_vector_type(4)))  bf16_t v4bf;
typedef __attribute__((ext_vector_type(8)))  bf16_t v8bf;
typedef __attribute__((ext_vector_type(16))) bf16_t v16bf;
typedef __attribute__((ext_vector_type(8)))  float  v8f;

// LDS tile stride: 32 k-columns + 8 pad = 40 bf16 = 80B (multiple of 16B, and
// 20 dwords -> the 16 lanes of a fragment read hit 16 disjoint 4-bank groups).
#define LDS_STRIDE 40

// ---------------------------------------------------------------------------
// CDNA5 async global->LDS path (ASYNCcnt) via inline asm (builtin prototypes
// differ across toolchains); synchronous fallback for non-gfx1250 passes.
// ---------------------------------------------------------------------------
static __device__ inline void async_copy16(const void* g, void* l) {
#if defined(__gfx1250__)
    unsigned lds_off = (unsigned)(unsigned long long)l;  // LDS offset = addr[31:0]
    asm volatile("global_load_async_to_lds_b128 %0, %1, off"
                 :
                 : "v"(lds_off), "v"(g)
                 : "memory");
#else
    *(uint4*)l = *(const uint4*)g;
#endif
}

template <int N>
static __device__ inline void async_waitcnt() {
#if defined(__gfx1250__)
    asm volatile("s_wait_asynccnt %0" : : "n"(N) : "memory");
#endif
}

// ---------------------------------------------------------------------------
// Helpers
// ---------------------------------------------------------------------------
static __device__ inline bf16_t f2bf(float f) {
    unsigned u = __builtin_bit_cast(unsigned, f);
    unsigned r = (u + 0x7FFFu + ((u >> 16) & 1u)) >> 16;
    unsigned short s = (unsigned short)r;
    return __builtin_bit_cast(bf16_t, s);
}

static __device__ inline float softplusf(float x) {
    return (x > 20.f) ? x : log1pf(__expf(x));
}

// Issue async copies for a 128x32 bf16 A-tile (row-major, leading dim ld)
// into LDS [128][LDS_STRIDE].  2 async b128 issues per thread.
static __device__ inline void issueA(const bf16_t* __restrict__ g, int ld,
                                     int rowBase, int kBase,
                                     bf16_t* __restrict__ lds) {
    const int tid = threadIdx.x;
#pragma unroll
    for (int i = 0; i < 2; ++i) {
        int ch  = tid + i * 256;      // 512 chunks of 16B
        int row = ch >> 2;            // 0..127
        int kc  = (ch & 3) * 8;       // 0,8,16,24
        async_copy16(g + (size_t)(rowBase + row) * ld + kBase + kc,
                     lds + row * LDS_STRIDE + kc);
    }
}

// Issue async copies for a 64x32 bf16 B-tile from a PRE-TRANSPOSED weight
// (WT is [N][K] row-major, leading dim ldK) into LDS [64][LDS_STRIDE].
// 1 async b128 issue per thread.
static __device__ inline void issueB(const bf16_t* __restrict__ gT, int ldK,
                                     int nBase, int kBase,
                                     bf16_t* __restrict__ lds) {
    const int tid = threadIdx.x;      // 256 chunks of 16B
    int n  = tid >> 2;                // 0..63
    int kc = (tid & 3) * 8;           // 0,8,16,24
    async_copy16(gT + (size_t)(nBase + n) * ldK + kBase + kc,
                 lds + n * LDS_STRIDE + kc);
}

// A fragment (16x32 bf16, ISA 7.12.2): lane<16 -> k 0..7 & 16..23 of row M=lane;
// lane>=16 -> k 8..15 & 24..31.  Two aligned b128 LDS loads.
static __device__ inline v16bf read_fragA(const bf16_t* __restrict__ lds,
                                          int wm, int f, int lane) {
    int row = wm * 32 + f * 16 + (lane & 15);
    int klo = (lane >> 4) * 8;
    const bf16_t* p = lds + row * LDS_STRIDE + klo;
    v8bf lo = *(const v8bf*)p;
    v8bf hi = *(const v8bf*)(p + 16);
    return __builtin_shufflevector(lo, hi, 0, 1, 2, 3, 4, 5, 6, 7,
                                           8, 9, 10, 11, 12, 13, 14, 15);
}

// B fragment (32x16 bf16): lanes 0..15 hold k 0..7 & 16..23 of column N=lane,
// lanes 16..31 hold k 8..15 & 24..31.  LDS is [n][k]; two aligned b128 loads.
static __device__ inline v16bf read_fragB(const bf16_t* __restrict__ lds,
                                          int wn, int g, int lane) {
    int n  = wn * 32 + g * 16 + (lane & 15);
    int k0 = (lane >> 4) * 8;
    const bf16_t* p = lds + n * LDS_STRIDE + k0;
    v8bf lo = *(const v8bf*)p;
    v8bf hi = *(const v8bf*)(p + 16);
    return __builtin_shufflevector(lo, hi, 0, 1, 2, 3, 4, 5, 6, 7,
                                           8, 9, 10, 11, 12, 13, 14, 15);
}

static __device__ inline v8f wmma_bf16(v16bf a, v16bf b, v8f c) {
    return __builtin_amdgcn_wmma_f32_16x16x32_bf16(
        false, a, false, b, (short)0, c, false, false);
}

// ---------------------------------------------------------------------------
// Stage 1a: plain f32 -> bf16 conversion (activations; n multiple of 1024)
// ---------------------------------------------------------------------------
__global__ __launch_bounds__(256) void cvt_f32_bf16(const float* __restrict__ src,
                                                    bf16_t* __restrict__ dst,
                                                    long n4) {
    long i = (long)blockIdx.x * 256 + threadIdx.x;
    if (i < n4) {
        float4 v = ((const float4*)src)[i];
        v4bf o;
        o[0] = f2bf(v.x); o[1] = f2bf(v.y); o[2] = f2bf(v.z); o[3] = f2bf(v.w);
        *(v4bf*)(dst + i * 4) = o;
    }
}

// ---------------------------------------------------------------------------
// Stage 1b: f32 [K][N] -> bf16 transposed [N][K] (weights), LDS-tiled 32x32.
// ---------------------------------------------------------------------------
__global__ __launch_bounds__(256) void cvt_transpose(const float* __restrict__ W,
                                                     bf16_t* __restrict__ WT,
                                                     int K, int N) {
    __shared__ float tile[32][33];
    const int kb = blockIdx.x * 32, nb = blockIdx.y * 32;
    const int tid = threadIdx.x;
    const int r = tid >> 3, c0 = (tid & 7) * 4;
    float4 v = *(const float4*)(W + (size_t)(kb + r) * N + nb + c0);
    tile[r][c0 + 0] = v.x; tile[r][c0 + 1] = v.y;
    tile[r][c0 + 2] = v.z; tile[r][c0 + 3] = v.w;
    __syncthreads();
    v4bf o;
#pragma unroll
    for (int j = 0; j < 4; ++j) o[j] = f2bf(tile[c0 + j][r]);
    *(v4bf*)(WT + (size_t)(nb + r) * K + kb + c0) = o;
}

// ---------------------------------------------------------------------------
// Stage 2: gates.  decay = exp(-step * exp(log_decay)) * carry_scale,
//   step = softplus(x@W_step + b_step) + 0.1*softplus(c@W_cstep)
// Two f32 accumulators share one K=D loop.  NT = 2+2+1+1 = 6 async issues.
// ---------------------------------------------------------------------------
__global__ __launch_bounds__(256) void gate_kernel(
    const bf16_t* __restrict__ xb, const bf16_t* __restrict__ cb,
    const bf16_t* __restrict__ WstepT, const bf16_t* __restrict__ WcstepT,
    const float* __restrict__ b_step, const float* __restrict__ log_decay,
    const float* __restrict__ carry_scale, bf16_t* __restrict__ decay_out) {
    __shared__ bf16_t ldsX[2][128 * LDS_STRIDE];
    __shared__ bf16_t ldsC[2][128 * LDS_STRIDE];
    __shared__ bf16_t ldsWs[2][64 * LDS_STRIDE];
    __shared__ bf16_t ldsWc[2][64 * LDS_STRIDE];

    const int nBase = blockIdx.x * 64;
    const int mBase = blockIdx.y * 128;
    const int tid = threadIdx.x, lane = tid & 31, wave = tid >> 5;
    const int wm = wave & 3, wn = wave >> 2;

    v8f acc1[2][2], acc2[2][2];
    const v8f vz = {0.f, 0.f, 0.f, 0.f, 0.f, 0.f, 0.f, 0.f};
#pragma unroll
    for (int f = 0; f < 2; ++f)
#pragma unroll
        for (int g = 0; g < 2; ++g) { acc1[f][g] = vz; acc2[f][g] = vz; }

    auto issue_group = [&](int kt, int b) {
        issueA(xb, DD, mBase, kt * 32, ldsX[b]);
        issueA(cb, DD, mBase, kt * 32, ldsC[b]);
        issueB(WstepT, DD, nBase, kt * 32, ldsWs[b]);
        issueB(WcstepT, DD, nBase, kt * 32, ldsWc[b]);
    };

    const int KT = DD / 32;
    issue_group(0, 0);
    for (int kt = 0; kt < KT; ++kt) {
        const int cur = kt & 1;
        if (kt + 1 < KT) { issue_group(kt + 1, cur ^ 1); async_waitcnt<6>(); }
        else             { async_waitcnt<0>(); }
        __syncthreads();

        v16bf ax[2], ac[2], bs[2], bc[2];
#pragma unroll
        for (int f = 0; f < 2; ++f) {
            ax[f] = read_fragA(ldsX[cur], wm, f, lane);
            ac[f] = read_fragA(ldsC[cur], wm, f, lane);
        }
#pragma unroll
        for (int g = 0; g < 2; ++g) {
            bs[g] = read_fragB(ldsWs[cur], wn, g, lane);
            bc[g] = read_fragB(ldsWc[cur], wn, g, lane);
        }
#pragma unroll
        for (int f = 0; f < 2; ++f)
#pragma unroll
            for (int g = 0; g < 2; ++g) {
                acc1[f][g] = wmma_bf16(ax[f], bs[g], acc1[f][g]);
                acc2[f][g] = wmma_bf16(ac[f], bc[g], acc2[f][g]);
            }
        __syncthreads();
    }

#pragma unroll
    for (int f = 0; f < 2; ++f)
#pragma unroll
        for (int g = 0; g < 2; ++g)
#pragma unroll
            for (int v = 0; v < 8; ++v) {
                int row = mBase + wm * 32 + f * 16 + (lane >> 4) * 8 + v;
                int col = nBase + wn * 32 + g * 16 + (lane & 15);
                float step = softplusf(acc1[f][g][v] + b_step[col]) +
                             0.1f * softplusf(acc2[f][g][v]);
                size_t idx = (size_t)row * SS + col;
                float dec = __expf(-step * __expf(log_decay[col])) * carry_scale[idx];
                decay_out[idx] = f2bf(dec);
            }
}

// ---------------------------------------------------------------------------
// Stage 3: proposal + state update.
//   proposal  = tanh(x@W_in + b_in + c@W_cin + state@W_state)
//   new_state = decay*state + (1-decay)*proposal
// Phase 1 (K=D) NT=6; phase 2 (K=S) NT=3.
// ---------------------------------------------------------------------------
__global__ __launch_bounds__(256) void prop_kernel(
    const bf16_t* __restrict__ xb, const bf16_t* __restrict__ cb,
    const bf16_t* __restrict__ sb,
    const bf16_t* __restrict__ WinT, const bf16_t* __restrict__ WcinT,
    const bf16_t* __restrict__ WstateT,
    const float* __restrict__ b_in, const bf16_t* __restrict__ decay_in,
    const float* __restrict__ state_f32,
    float* __restrict__ new_state_out, bf16_t* __restrict__ nsb_out) {
    __shared__ bf16_t ldsX[2][128 * LDS_STRIDE];
    __shared__ bf16_t ldsC[2][128 * LDS_STRIDE];
    __shared__ bf16_t ldsWa[2][64 * LDS_STRIDE];
    __shared__ bf16_t ldsWb[2][64 * LDS_STRIDE];

    const int nBase = blockIdx.x * 64;
    const int mBase = blockIdx.y * 128;
    const int tid = threadIdx.x, lane = tid & 31, wave = tid >> 5;
    const int wm = wave & 3, wn = wave >> 2;

    v8f acc[2][2];
    const v8f vz = {0.f, 0.f, 0.f, 0.f, 0.f, 0.f, 0.f, 0.f};
#pragma unroll
    for (int f = 0; f < 2; ++f)
#pragma unroll
        for (int g = 0; g < 2; ++g) acc[f][g] = vz;

    // ---- Phase 1: K = D, acc += x@W_in + c@W_cin
    auto issue1 = [&](int kt, int b) {
        issueA(xb, DD, mBase, kt * 32, ldsX[b]);
        issueA(cb, DD, mBase, kt * 32, ldsC[b]);
        issueB(WinT, DD, nBase, kt * 32, ldsWa[b]);
        issueB(WcinT, DD, nBase, kt * 32, ldsWb[b]);
    };
    const int KT1 = DD / 32;
    issue1(0, 0);
    for (int kt = 0; kt < KT1; ++kt) {
        const int cur = kt & 1;
        if (kt + 1 < KT1) { issue1(kt + 1, cur ^ 1); async_waitcnt<6>(); }
        else              { async_waitcnt<0>(); }
        __syncthreads();

        v16bf ax[2], ac[2], ba[2], bb[2];
#pragma unroll
        for (int f = 0; f < 2; ++f) {
            ax[f] = read_fragA(ldsX[cur], wm, f, lane);
            ac[f] = read_fragA(ldsC[cur], wm, f, lane);
        }
#pragma unroll
        for (int g = 0; g < 2; ++g) {
            ba[g] = read_fragB(ldsWa[cur], wn, g, lane);
            bb[g] = read_fragB(ldsWb[cur], wn, g, lane);
        }
#pragma unroll
        for (int f = 0; f < 2; ++f)
#pragma unroll
            for (int g = 0; g < 2; ++g) {
                acc[f][g] = wmma_bf16(ax[f], ba[g], acc[f][g]);
                acc[f][g] = wmma_bf16(ac[f], bb[g], acc[f][g]);
            }
        __syncthreads();
    }

    // ---- Phase 2: K = S, acc += state@W_state
    auto issue2 = [&](int kt, int b) {
        issueA(sb, SS, mBase, kt * 32, ldsX[b]);
        issueB(WstateT, SS, nBase, kt * 32, ldsWa[b]);
    };
    const int KT2 = SS / 32;
    issue2(0, 0);
    for (int kt = 0; kt < KT2; ++kt) {
        const int cur = kt & 1;
        if (kt + 1 < KT2) { issue2(kt + 1, cur ^ 1); async_waitcnt<3>(); }
        else              { async_waitcnt<0>(); }
        __syncthreads();

        v16bf as[2], bw[2];
#pragma unroll
        for (int f = 0; f < 2; ++f) as[f] = read_fragA(ldsX[cur], wm, f, lane);
#pragma unroll
        for (int g = 0; g < 2; ++g) bw[g] = read_fragB(ldsWa[cur], wn, g, lane);
#pragma unroll
        for (int f = 0; f < 2; ++f)
#pragma unroll
            for (int g = 0; g < 2; ++g)
                acc[f][g] = wmma_bf16(as[f], bw[g], acc[f][g]);
        __syncthreads();
    }

#pragma unroll
    for (int f = 0; f < 2; ++f)
#pragma unroll
        for (int g = 0; g < 2; ++g)
#pragma unroll
            for (int v = 0; v < 8; ++v) {
                int row = mBase + wm * 32 + f * 16 + (lane >> 4) * 8 + v;
                int col = nBase + wn * 32 + g * 16 + (lane & 15);
                size_t idx = (size_t)row * SS + col;
                float prop = tanhf(acc[f][g][v] + b_in[col]);
                float dec  = (float)decay_in[idx];
                float ns   = dec * state_f32[idx] + (1.f - dec) * prop;
                new_state_out[idx] = ns;
                nsb_out[idx] = f2bf(ns);
            }
}

// ---------------------------------------------------------------------------
// Stage 4: h = concat([x, new_state]) @ W_out + b_out + x.  NT = 3.
// ---------------------------------------------------------------------------
__global__ __launch_bounds__(256) void out_kernel(
    const bf16_t* __restrict__ xb, const bf16_t* __restrict__ nsb,
    const bf16_t* __restrict__ WoutT, const float* __restrict__ b_out,
    const float* __restrict__ x_f32, float* __restrict__ h_out) {
    __shared__ bf16_t ldsA[2][128 * LDS_STRIDE];
    __shared__ bf16_t ldsW[2][64 * LDS_STRIDE];

    const int nBase = blockIdx.x * 64;
    const int mBase = blockIdx.y * 128;
    const int tid = threadIdx.x, lane = tid & 31, wave = tid >> 5;
    const int wm = wave & 3, wn = wave >> 2;

    v8f acc[2][2];
    const v8f vz = {0.f, 0.f, 0.f, 0.f, 0.f, 0.f, 0.f, 0.f};
#pragma unroll
    for (int f = 0; f < 2; ++f)
#pragma unroll
        for (int g = 0; g < 2; ++g) acc[f][g] = vz;

    auto issueOut = [&](int kt, int b) {
        const bf16_t* a = (kt < DD / 32) ? xb : nsb;
        int ld    = (kt < DD / 32) ? DD : SS;
        int kB    = (kt < DD / 32) ? kt * 32 : kt * 32 - DD;
        issueA(a, ld, mBase, kB, ldsA[b]);
        issueB(WoutT, DD + SS, nBase, kt * 32, ldsW[b]);
    };
    const int KT = (DD + SS) / 32;   // 96
    issueOut(0, 0);
    for (int kt = 0; kt < KT; ++kt) {
        const int cur = kt & 1;
        if (kt + 1 < KT) { issueOut(kt + 1, cur ^ 1); async_waitcnt<3>(); }
        else             { async_waitcnt<0>(); }
        __syncthreads();

        v16bf af[2], bw[2];
#pragma unroll
        for (int f = 0; f < 2; ++f) af[f] = read_fragA(ldsA[cur], wm, f, lane);
#pragma unroll
        for (int g = 0; g < 2; ++g) bw[g] = read_fragB(ldsW[cur], wn, g, lane);
#pragma unroll
        for (int f = 0; f < 2; ++f)
#pragma unroll
            for (int g = 0; g < 2; ++g)
                acc[f][g] = wmma_bf16(af[f], bw[g], acc[f][g]);
        __syncthreads();
    }

#pragma unroll
    for (int f = 0; f < 2; ++f)
#pragma unroll
        for (int g = 0; g < 2; ++g)
#pragma unroll
            for (int v = 0; v < 8; ++v) {
                int row = mBase + wm * 32 + f * 16 + (lane >> 4) * 8 + v;
                int col = nBase + wn * 32 + g * 16 + (lane & 15);
                size_t idx = (size_t)row * DD + col;
                h_out[idx] = acc[f][g][v] + b_out[col] + x_f32[idx];
            }
}

// ---------------------------------------------------------------------------
// Stage 5: row LayerNorm over D=1024 (one block per row, 4 elems/thread)
// ---------------------------------------------------------------------------
__global__ __launch_bounds__(256) void ln_kernel(const float* __restrict__ h,
                                                 const float* __restrict__ ln_w,
                                                 const float* __restrict__ ln_b,
                                                 float* __restrict__ out) {
    __shared__ float s1[256], s2[256];
    const int row = blockIdx.x, tid = threadIdx.x;
    const float* hr = h + (size_t)row * DD;
    float4 v = ((const float4*)hr)[tid];
    s1[tid] = v.x + v.y + v.z + v.w;
    s2[tid] = v.x * v.x + v.y * v.y + v.z * v.z + v.w * v.w;
    __syncthreads();
#pragma unroll
    for (int s = 128; s > 0; s >>= 1) {
        if (tid < s) { s1[tid] += s1[tid + s]; s2[tid] += s2[tid + s]; }
        __syncthreads();
    }
    float mu   = s1[0] * (1.f / DD);
    float var  = s2[0] * (1.f / DD) - mu * mu;
    float rstd = rsqrtf(var + LN_EPS);
    float4 w = ((const float4*)ln_w)[tid];
    float4 b = ((const float4*)ln_b)[tid];
    float4 o;
    o.x = (v.x - mu) * rstd * w.x + b.x;
    o.y = (v.y - mu) * rstd * w.y + b.y;
    o.z = (v.z - mu) * rstd * w.z + b.z;
    o.w = (v.w - mu) * rstd * w.w + b.w;
    ((float4*)(out + (size_t)row * DD))[tid] = o;
}

// ---------------------------------------------------------------------------
// Host launcher
// ---------------------------------------------------------------------------
extern "C" void kernel_launch(void* const* d_in, const int* in_sizes, int n_in,
                              void* d_out, int out_size, void* d_ws, size_t ws_size,
                              hipStream_t stream) {
    (void)in_sizes; (void)n_in; (void)out_size; (void)ws_size;

    const float* x_t       = (const float*)d_in[0];
    const float* state     = (const float*)d_in[1];
    const float* cond      = (const float*)d_in[2];
    const float* carry     = (const float*)d_in[3];
    const float* W_step    = (const float*)d_in[4];
    const float* b_step    = (const float*)d_in[5];
    const float* W_cstep   = (const float*)d_in[6];
    const float* W_in      = (const float*)d_in[7];
    const float* b_in      = (const float*)d_in[8];
    const float* W_cin     = (const float*)d_in[9];
    const float* W_state   = (const float*)d_in[10];
    const float* W_out     = (const float*)d_in[11];
    const float* b_out     = (const float*)d_in[12];
    const float* ln_w      = (const float*)d_in[13];
    const float* ln_b      = (const float*)d_in[14];
    const float* log_decay = (const float*)d_in[15];

    float* out_ln    = (float*)d_out;                       // (B, D)
    float* out_state = (float*)d_out + (size_t)BB * DD;     // (B, S)

    // Workspace carve-up
    char* ws = (char*)d_ws;
    size_t off = 0;
    auto take = [&](size_t bytes) { char* p = ws + off; off += bytes; return p; };
    bf16_t* xb      = (bf16_t*)take((size_t)BB * DD * 2);
    bf16_t* cb      = (bf16_t*)take((size_t)BB * DD * 2);
    bf16_t* sb      = (bf16_t*)take((size_t)BB * SS * 2);
    bf16_t* nsb     = (bf16_t*)take((size_t)BB * SS * 2);
    bf16_t* decayb  = (bf16_t*)take((size_t)BB * SS * 2);
    bf16_t* WstepT  = (bf16_t*)take((size_t)DD * SS * 2);   // [S][D]
    bf16_t* WcstepT = (bf16_t*)take((size_t)DD * SS * 2);
    bf16_t* WinT    = (bf16_t*)take((size_t)DD * SS * 2);
    bf16_t* WcinT   = (bf16_t*)take((size_t)DD * SS * 2);
    bf16_t* WstateT = (bf16_t*)take((size_t)SS * SS * 2);   // [S][S]
    bf16_t* WoutT   = (bf16_t*)take((size_t)(DD + SS) * DD * 2); // [D][D+S]
    float*  hbuf    = (float*)take((size_t)BB * DD * 4);

    auto cvt = [&](const float* s, bf16_t* d, long n) {
        long n4 = n / 4;
        cvt_f32_bf16<<<(n4 + 255) / 256, 256, 0, stream>>>(s, d, n4);
    };
    auto cvtT = [&](const float* s, bf16_t* d, int K, int N) {
        cvt_transpose<<<dim3(K / 32, N / 32), 256, 0, stream>>>(s, d, K, N);
    };

    // Stage 1: conversions (+ weight transposes)
    cvt(x_t,   xb, (long)BB * DD);
    cvt(cond,  cb, (long)BB * DD);
    cvt(state, sb, (long)BB * SS);
    cvtT(W_step,  WstepT,  DD, SS);
    cvtT(W_cstep, WcstepT, DD, SS);
    cvtT(W_in,    WinT,    DD, SS);
    cvtT(W_cin,   WcinT,   DD, SS);
    cvtT(W_state, WstateT, SS, SS);
    cvtT(W_out,   WoutT,   DD + SS, DD);

    dim3 blk(256);
    gate_kernel<<<dim3(SS / 64, BB / 128), blk, 0, stream>>>(
        xb, cb, WstepT, WcstepT, b_step, log_decay, carry, decayb);
    prop_kernel<<<dim3(SS / 64, BB / 128), blk, 0, stream>>>(
        xb, cb, sb, WinT, WcinT, WstateT, b_in, decayb, state, out_state, nsb);
    out_kernel<<<dim3(DD / 64, BB / 128), blk, 0, stream>>>(
        xb, nsb, WoutT, b_out, x_t, hbuf);
    ln_kernel<<<BB, blk, 0, stream>>>(hbuf, ln_w, ln_b, out_ln);
}
